// W2NER_35948876268306
// MI455X (gfx1250) — compile-verified, matrix operand order checked by reading, not run on hardware
//
#include <hip/hip_runtime.h>
#include <hip/hip_bf16.h>
#include <math.h>
#include <stdint.h>

typedef __bf16 bf16;
typedef __attribute__((ext_vector_type(16))) __bf16 v16bf;
typedef __attribute__((ext_vector_type(8)))  __bf16 v8bf;
typedef __attribute__((ext_vector_type(8)))  float  v8f;

#define BB    4
#define NN    192
#define BN    768      // B*N
#define DIN   768
#define HDIM  256
#define G4    1024     // 4*HD
#define HHX   512      // H
#define PPX   513      // BIAFF+1
#define PADK  544      // 513 padded to 17*32
#define CLSN  10
#define UW    (CLSN*PADK)   // 5440
#define WCW   1051     // 2*513+25
#define NPOS  30

__device__ __forceinline__ bf16 f2bf(float f) {
  union { float f; uint32_t u; } in; in.f = f;
  uint32_t u = in.u;
  uint32_t lsb = (u >> 16) & 1u;
  u += 0x7FFFu + lsb;
  union { uint16_t s; bf16 b; } out; out.s = (uint16_t)(u >> 16);
  return out.b;
}
__device__ __forceinline__ float sigm(float x) { return 1.0f / (1.0f + expf(-x)); }

// ---------------------------------------------------------------------------
// WMMA GEMM, wave-per-16x64-strip, software-pipelined:
//   C(MxN) = act(A(MxK) * B + bias)
// A row-major bf16; B supplied TRANSPOSED (BT is NxK row-major bf16) so both
// fragments load as two contiguous 16B vectors per lane. One A fragment feeds
// 4 v_wmma calls per k-step. Fragments for step k+1 are loaded BEFORE the
// WMMAs of step k (double-buffered registers), so each WMMA waits only on
// loads issued a full iteration earlier.
// K % 32 == 0, M % 16 == 0, N % 64 == 0. Optional batching over blockIdx.y.
// ---------------------------------------------------------------------------
#define LOADF(dst, base, kk)                                   \
  dst.h[0] = *(const v8bf*)((base) + (kk) + hh * 8);           \
  dst.h[1] = *(const v8bf*)((base) + (kk) + 16 + hh * 8)

__global__ __launch_bounds__(256) void k_gemm(
    const bf16* __restrict__ A, const bf16* __restrict__ BT,
    const float* __restrict__ bias, float* __restrict__ Cf, bf16* __restrict__ Cb,
    int M, int N, int K, int lda, int ldbt, int ldcf, int ldcb, int act,
    int divY, long long sA1, long long sA2, long long sB1, long long sC1)
{
  int y = blockIdx.y;
  A  += (long long)(y / divY) * sA1 + (long long)(y % divY) * sA2;
  BT += (long long)(y / divY) * sB1;
  long long coff = (long long)y * sC1;

  int ntn = N >> 6;                       // 64-wide strips
  int tiles = (M >> 4) * ntn;
  int wid = blockIdx.x * 8 + (threadIdx.x >> 5);
  if (wid >= tiles) return;
  int tm = wid / ntn, tn = wid % ntn;
  int lane = threadIdx.x & 31;
  int r = lane & 15, hh = lane >> 4;

  const bf16* arow = A + (long long)(tm * 16 + r) * lda;
  const bf16* brow0 = BT + (long long)(tn * 64 +  0 + r) * ldbt;
  const bf16* brow1 = BT + (long long)(tn * 64 + 16 + r) * ldbt;
  const bf16* brow2 = BT + (long long)(tn * 64 + 32 + r) * ldbt;
  const bf16* brow3 = BT + (long long)(tn * 64 + 48 + r) * ldbt;

  v8f acc0 = (v8f){0.f,0.f,0.f,0.f,0.f,0.f,0.f,0.f};
  v8f acc1 = acc0, acc2 = acc0, acc3 = acc0;

  union Frag { v16bf v; v8bf h[2]; };
  // prologue: load k=0 fragments
  Frag a, b0, b1, b2, b3;
  LOADF(a, arow, 0);
  LOADF(b0, brow0, 0); LOADF(b1, brow1, 0);
  LOADF(b2, brow2, 0); LOADF(b3, brow3, 0);

  for (int k0 = 0; k0 < K; k0 += 32) {
    // next-iteration index; clamp to 0 on the last trip (valid, unused load)
    int kn = (k0 + 32 < K) ? (k0 + 32) : 0;
    if (k0 + 64 < K) {                    // prefetch one block further ahead
      __builtin_prefetch(arow + k0 + 64, 0, 3);
      __builtin_prefetch(brow0 + k0 + 64, 0, 3);
    }
    // ---- load phase for step k+1 (fresh registers, kept live) ----
    Frag an, bn0, bn1, bn2, bn3;
    LOADF(an, arow, kn);
    LOADF(bn0, brow0, kn); LOADF(bn1, brow1, kn);
    LOADF(bn2, brow2, kn); LOADF(bn3, brow3, kn);
    // ---- compute phase for step k: A reused 4x from registers ----
    acc0 = __builtin_amdgcn_wmma_f32_16x16x32_bf16(false, a.v, false, b0.v,
                                                   (short)0, acc0, false, false);
    acc1 = __builtin_amdgcn_wmma_f32_16x16x32_bf16(false, a.v, false, b1.v,
                                                   (short)0, acc1, false, false);
    acc2 = __builtin_amdgcn_wmma_f32_16x16x32_bf16(false, a.v, false, b2.v,
                                                   (short)0, acc2, false, false);
    acc3 = __builtin_amdgcn_wmma_f32_16x16x32_bf16(false, a.v, false, b3.v,
                                                   (short)0, acc3, false, false);
    // ---- rotate ----
    a = an; b0 = bn0; b1 = bn1; b2 = bn2; b3 = bn3;
  }

  v8f accs[4] = {acc0, acc1, acc2, acc3};
  #pragma unroll
  for (int s = 0; s < 4; ++s) {
    int col = tn * 64 + s * 16 + r;
    float bv = bias ? bias[col] : 0.f;
    #pragma unroll
    for (int v = 0; v < 8; ++v) {
      int row = tm * 16 + v + hh * 8;   // C/D layout: VGPR v -> M = v (+8 for hi lanes)
      float val = accs[s][v] + bv;
      if (act == 1) val = 0.5f * val * (1.0f + erff(val * 0.70710678f));   // exact GELU
      else if (act == 2) val = val > 0.f ? val : 0.01f * val;              // leaky relu
      if (Cf) Cf[coff + (long long)row * ldcf + col] = val;
      if (Cb) Cb[coff + (long long)row * ldcb + col] = f2bf(val);
    }
  }
}

// ---------------------------------------------------------------------------
// Elementwise prep kernels
// ---------------------------------------------------------------------------
__global__ void k_f2bf(const float* __restrict__ src, bf16* __restrict__ dst, int n) {
  int id = blockIdx.x * blockDim.x + threadIdx.x;
  if (id < n) dst[id] = f2bf(src[id]);
}

// x_rev[b,t,:] = word_reps[b, clip(wl[b]-1-t,0,N-1), :]  (cast to bf16)
__global__ void k_xrev(const float* __restrict__ wr_in, const int* __restrict__ wl,
                       bf16* __restrict__ dst) {
  int id = blockIdx.x * blockDim.x + threadIdx.x;
  if (id >= BN * DIN) return;
  int d = id % DIN; int row = id / DIN; int t = row % NN; int b = row / NN;
  int idx = wl[b] - 1 - t;
  idx = idx < 0 ? 0 : (idx > NN - 1 ? NN - 1 : idx);
  dst[id] = f2bf(wr_in[((long long)(b * NN + idx)) * DIN + d]);
}

// Whh (1024x256) -> WhhT (256x1024) for coalesced recurrent reads
__global__ void k_whhT(const float* __restrict__ src, float* __restrict__ dst) {
  int id = blockIdx.x * blockDim.x + threadIdx.x;
  if (id >= G4 * HDIM) return;
  int g = id % G4; int k = id / G4;
  dst[id] = src[g * HDIM + k];
}

// pad columns of h1_pad/t1_pad: col 512 = 1 (the "append ones"), 513..543 = 0
__global__ void k_pads(bf16* __restrict__ h1, bf16* __restrict__ t1) {
  int id = blockIdx.x * blockDim.x + threadIdx.x;
  if (id >= 2 * BN * 32) return;
  int cc = id % 32; int row = (id / 32) % BN; int which = id / (32 * BN);
  bf16* p = which ? t1 : h1;
  p[(long long)row * PADK + 512 + cc] = f2bf(cc == 0 ? 1.0f : 0.0f);
}

// WcatT[o*544+j][i] = biaff_W[o,i,j]  (zero padded), bf16
__global__ void k_wcat(const float* __restrict__ W, bf16* __restrict__ out) {
  long long id = (long long)blockIdx.x * blockDim.x + threadIdx.x;
  if (id >= (long long)CLSN * PADK * PADK) return;
  int i = (int)(id % PADK); long long rr = id / PADK;
  int j = (int)(rr % PADK); int o = (int)(rr / PADK);
  float v = (i < PPX && j < PPX) ? W[((long long)o * PPX + i) * PPX + j] : 0.f;
  out[((long long)o * PADK + j) * PADK + i] = f2bf(v);
}

// ---------------------------------------------------------------------------
// Sequential BiLSTM: block 0 = forward, block 1 = backward (on reversed input)
// h/c/gates in LDS; xw precomputed by WMMA GEMM; WhhT coalesced.
// ---------------------------------------------------------------------------
__global__ __launch_bounds__(1024) void k_lstm(
    const float* __restrict__ xw_f, const float* __restrict__ xw_b,
    const float* __restrict__ WhhT_f, const float* __restrict__ WhhT_b,
    const int* __restrict__ wl, float* __restrict__ wr, float* __restrict__ bwd_rev)
{
  int dir = blockIdx.x;
  const float* xw   = dir ? xw_b : xw_f;
  const float* WhhT = dir ? WhhT_b : WhhT_f;
  __shared__ float sh[BB][HDIM], sc[BB][HDIM], sg[BB][G4];
  int tid = threadIdx.x;
  int bme = tid >> 8, jme = tid & 255;
  sh[bme][jme] = 0.f; sc[bme][jme] = 0.f;
  __syncthreads();
  int g = tid;
  for (int t = 0; t < NN; ++t) {
    float a0 = xw[(long long)(0 * NN + t) * G4 + g];
    float a1 = xw[(long long)(1 * NN + t) * G4 + g];
    float a2 = xw[(long long)(2 * NN + t) * G4 + g];
    float a3 = xw[(long long)(3 * NN + t) * G4 + g];
    #pragma unroll 4
    for (int k = 0; k < HDIM; ++k) {
      float wk = WhhT[k * G4 + g];
      a0 += sh[0][k] * wk; a1 += sh[1][k] * wk;
      a2 += sh[2][k] * wk; a3 += sh[3][k] * wk;
    }
    sg[0][g] = a0; sg[1][g] = a1; sg[2][g] = a2; sg[3][g] = a3;
    __syncthreads();
    float ig = sg[bme][jme], fg = sg[bme][256 + jme];
    float gg = sg[bme][512 + jme], og = sg[bme][768 + jme];
    float cn = sigm(fg) * sc[bme][jme] + sigm(ig) * tanhf(gg);
    float hn = sigm(og) * tanhf(cn);
    bool m = t < wl[bme];
    float out = m ? hn : 0.f;
    if (dir == 0) wr[(long long)(bme * NN + t) * HHX + jme] = out;
    else          bwd_rev[(long long)(bme * NN + t) * HDIM + jme] = out;
    if (m) { sh[bme][jme] = hn; sc[bme][jme] = cn; }
    __syncthreads();
  }
  if (dir == 1) {
    __threadfence_block();
    for (int t = 0; t < NN; ++t) {
      int L = wl[bme];
      int idx = L - 1 - t; idx = idx < 0 ? 0 : (idx > NN - 1 ? NN - 1 : idx);
      float v = (t < L) ? bwd_rev[(long long)(bme * NN + idx) * HDIM + jme] : 0.f;
      wr[(long long)(bme * NN + t) * HHX + HDIM + jme] = v;
    }
  }
}

// ---------------------------------------------------------------------------
// o3 pieces
// ---------------------------------------------------------------------------
__global__ void k_terms(const float* __restrict__ head_f, const float* __restrict__ tail_f,
                        const float* __restrict__ W_co,
                        float* __restrict__ term1, float* __restrict__ term2) {
  int id = blockIdx.x * blockDim.x + threadIdx.x;
  int total = BB * CLSN * NN;
  if (id >= 2 * total) return;
  int kind = id / total; int rem = id % total;
  int x = rem % NN; int bo = rem / NN; int o = bo % CLSN; int b = bo / CLSN;
  const float* src = (kind ? tail_f : head_f) + (long long)(b * NN + x) * HHX;
  const float* wco = W_co + (long long)o * WCW + (kind ? PPX : 0);
  float s = wco[HHX];                    // "ones" coefficient (index 512)
  for (int i = 0; i < HHX; ++i) s += src[i] * wco[i];
  (kind ? term2 : term1)[bo * NN + x] = s;
}

// table[o][span_idx] = dot(size_emb[span_idx], W_co[o, 1026:1051])
__global__ void k_table(const float* __restrict__ size_emb, const float* __restrict__ W_co,
                        float* __restrict__ table) {
  int id = blockIdx.x * blockDim.x + threadIdx.x;
  if (id >= CLSN * NPOS) return;
  int o = id / NPOS, s = id % NPOS;
  float acc = 0.f;
  for (int e = 0; e < 25; ++e) acc += size_emb[s * 25 + e] * W_co[(long long)o * WCW + 1026 + e];
  table[id] = acc;
}

__global__ void k_final(float* __restrict__ out, const float* __restrict__ term1,
                        const float* __restrict__ term2, const float* __restrict__ table) {
  long long id = (long long)blockIdx.x * blockDim.x + threadIdx.x;
  const long long total = (long long)BB * CLSN * NN * NN;
  if (id >= total) return;
  int n = (int)(id % NN); long long r = id / NN;
  int m = (int)(r % NN); r /= NN;
  int o = (int)(r % CLSN); int b = (int)(r / CLSN);
  int d = n - m; d = d < -15 ? -15 : (d > 14 ? 14 : d);
  out[id] += term1[(b * CLSN + o) * NN + m] + term2[(b * CLSN + o) * NN + n]
           + table[o * NPOS + d + 15];
}

// ---------------------------------------------------------------------------
extern "C" void kernel_launch(void* const* d_in, const int* in_sizes, int n_in,
                              void* d_out, int out_size, void* d_ws, size_t ws_size,
                              hipStream_t stream) {
  (void)in_sizes; (void)n_in; (void)out_size; (void)ws_size;
  const float* word_reps = (const float*)d_in[0];
  const int*   wl        = (const int*)d_in[1];
  const float* Wih_f = (const float*)d_in[4];
  const float* Whh_f = (const float*)d_in[5];
  const float* b_f   = (const float*)d_in[6];
  const float* Wih_b = (const float*)d_in[7];
  const float* Whh_b = (const float*)d_in[8];
  const float* b_b   = (const float*)d_in[9];
  const float* mlp1_w = (const float*)d_in[20];
  const float* mlp1_b = (const float*)d_in[21];
  const float* mlp2_w = (const float*)d_in[22];
  const float* mlp2_b = (const float*)d_in[23];
  const float* head_w = (const float*)d_in[24];
  const float* head_b = (const float*)d_in[25];
  const float* tail_w = (const float*)d_in[26];
  const float* tail_b = (const float*)d_in[27];
  const float* biaffW = (const float*)d_in[28];
  const float* size_emb = (const float*)d_in[29];
  const float* W_co   = (const float*)d_in[30];
  float* out = (float*)d_out;

  // workspace carve (256B aligned)
  char* base = (char*)d_ws; size_t off = 0;
  auto carve = [&](size_t bytes) -> char* {
    off = (off + 255) & ~(size_t)255; char* p = base + off; off += bytes; return p;
  };
  bf16* wrin_bf = (bf16*)carve((size_t)BN * DIN * 2);
  bf16* xrev_bf = (bf16*)carve((size_t)BN * DIN * 2);
  bf16* wihf_bf = (bf16*)carve((size_t)G4 * DIN * 2);
  bf16* wihb_bf = (bf16*)carve((size_t)G4 * DIN * 2);
  bf16* m1_bf   = (bf16*)carve((size_t)HHX * HHX * 2);
  bf16* m2_bf   = (bf16*)carve((size_t)HHX * HHX * 2);
  bf16* hw_bf   = (bf16*)carve((size_t)HHX * HHX * 2);
  bf16* tw_bf   = (bf16*)carve((size_t)HHX * HHX * 2);
  float* whhTf  = (float*)carve((size_t)G4 * HDIM * 4);
  float* whhTb  = (float*)carve((size_t)G4 * HDIM * 4);
  float* xw_f   = (float*)carve((size_t)BN * G4 * 4);
  float* xw_b   = (float*)carve((size_t)BN * G4 * 4);
  float* wr     = (float*)carve((size_t)BN * HHX * 4);
  float* bwd_rev= (float*)carve((size_t)BN * HDIM * 4);
  bf16* wr_bf   = (bf16*)carve((size_t)BN * HHX * 2);
  bf16* h1_pad  = (bf16*)carve((size_t)BN * PADK * 2);
  bf16* t1_pad  = (bf16*)carve((size_t)BN * PADK * 2);
  float* head_f = (float*)carve((size_t)BN * HHX * 4);
  float* tail_f = (float*)carve((size_t)BN * HHX * 4);
  bf16* wcatT   = (bf16*)carve((size_t)UW * PADK * 2);
  bf16* U_bf    = (bf16*)carve((size_t)BN * UW * 2);
  float* term1  = (float*)carve((size_t)BB * CLSN * NN * 4);
  float* term2  = (float*)carve((size_t)BB * CLSN * NN * 4);
  float* table  = (float*)carve((size_t)CLSN * NPOS * 4);

  auto blks = [](long long n) { return (unsigned)((n + 255) / 256); };
  auto gemm = [&](const bf16* A, const bf16* BT, const float* bias, float* Cf, bf16* Cb,
                  int M, int Nn, int K, int lda, int ldbt, int ldcf, int ldcb, int act,
                  int batch, int divY, long long sA1, long long sA2, long long sB1,
                  long long sC1) {
    int tiles = (M / 16) * (Nn / 64);       // 16x64 strip per wave
    dim3 grid((tiles + 7) / 8, batch);
    k_gemm<<<grid, 256, 0, stream>>>(A, BT, bias, Cf, Cb, M, Nn, K, lda, ldbt,
                                     ldcf, ldcb, act, divY, sA1, sA2, sB1, sC1);
  };

  // --- prep: bf16 casts / transposes / padding / gather ---
  k_f2bf<<<blks((long long)BN * DIN), 256, 0, stream>>>(word_reps, wrin_bf, BN * DIN);
  k_xrev<<<blks((long long)BN * DIN), 256, 0, stream>>>(word_reps, wl, xrev_bf);
  k_f2bf<<<blks((long long)G4 * DIN), 256, 0, stream>>>(Wih_f, wihf_bf, G4 * DIN);
  k_f2bf<<<blks((long long)G4 * DIN), 256, 0, stream>>>(Wih_b, wihb_bf, G4 * DIN);
  k_f2bf<<<blks((long long)HHX * HHX), 256, 0, stream>>>(mlp1_w, m1_bf, HHX * HHX);
  k_f2bf<<<blks((long long)HHX * HHX), 256, 0, stream>>>(mlp2_w, m2_bf, HHX * HHX);
  k_f2bf<<<blks((long long)HHX * HHX), 256, 0, stream>>>(head_w, hw_bf, HHX * HHX);
  k_f2bf<<<blks((long long)HHX * HHX), 256, 0, stream>>>(tail_w, tw_bf, HHX * HHX);
  k_whhT<<<blks((long long)G4 * HDIM), 256, 0, stream>>>(Whh_f, whhTf);
  k_whhT<<<blks((long long)G4 * HDIM), 256, 0, stream>>>(Whh_b, whhTb);
  k_pads<<<blks((long long)2 * BN * 32), 256, 0, stream>>>(h1_pad, t1_pad);
  k_wcat<<<blks((long long)CLSN * PADK * PADK), 256, 0, stream>>>(biaffW, wcatT);

  // --- LSTM input projections (WMMA) ---
  gemm(wrin_bf, wihf_bf, b_f, xw_f, nullptr, BN, G4, DIN, DIN, DIN, G4, 0, 0,
       1, 1, 0, 0, 0, 0);
  gemm(xrev_bf, wihb_bf, b_b, xw_b, nullptr, BN, G4, DIN, DIN, DIN, G4, 0, 0,
       1, 1, 0, 0, 0, 0);

  // --- sequential recurrence, both directions concurrently ---
  k_lstm<<<2, 1024, 0, stream>>>(xw_f, xw_b, whhTf, whhTb, wl, wr, bwd_rev);
  k_f2bf<<<blks((long long)BN * HHX), 256, 0, stream>>>(wr, wr_bf, BN * HHX);

  // --- MLP heads (WMMA, fused bias+activation) ---
  gemm(wr_bf, m1_bf, mlp1_b, nullptr, h1_pad, BN, HHX, HHX, HHX, HHX, 0, PADK, 1,
       1, 1, 0, 0, 0, 0);                                   // GELU -> h1 (bf16, padded)
  gemm(wr_bf, m2_bf, mlp2_b, nullptr, t1_pad, BN, HHX, HHX, HHX, HHX, 0, PADK, 1,
       1, 1, 0, 0, 0, 0);                                   // GELU -> t1
  gemm(wr_bf, hw_bf, head_b, head_f, nullptr, BN, HHX, HHX, HHX, HHX, HHX, 0, 2,
       1, 1, 0, 0, 0, 0);                                   // leaky -> head
  gemm(wr_bf, tw_bf, tail_b, tail_f, nullptr, BN, HHX, HHX, HHX, HHX, HHX, 0, 2,
       1, 1, 0, 0, 0, 0);                                   // leaky -> tail

  // --- biaffine stage 1: U[x, o*544+j] = sum_i h1[x,i] W[o,i,j] ---
  gemm(h1_pad, wcatT, nullptr, nullptr, U_bf, BN, UW, PADK, PADK, PADK, 0, UW, 0,
       1, 1, 0, 0, 0, 0);

  // --- biaffine stage 2 (batched over 40 (b,o)): out[b,o,x,y] = U_bo @ t1_b^T ---
  gemm(U_bf, t1_pad, nullptr, out, nullptr, NN, NN, PADK, UW, PADK, NN, 0, 0,
       40, 10, (long long)NN * UW, (long long)PADK, (long long)NN * PADK,
       (long long)NN * NN);

  // --- o3 rank-1 terms + span table + final fused add ---
  k_terms<<<blks((long long)2 * BB * CLSN * NN), 256, 0, stream>>>(head_f, tail_f, W_co,
                                                                  term1, term2);
  k_table<<<blks((long long)CLSN * NPOS), 256, 0, stream>>>(size_emb, W_co, table);
  k_final<<<blks((long long)BB * CLSN * NN * NN), 256, 0, stream>>>(out, term1, term2,
                                                                   table);
}
